// GraphInteractionLayer_89584427860492
// MI455X (gfx1250) — matrix operand order repaired and checked by reading, not compiled, not run on hardware
//
#include <hip/hip_runtime.h>
#include <cstdint>

#define N_NODES 25000
#define N_EDGES 250000
#define DIM 128

typedef __attribute__((ext_vector_type(16))) __bf16 v16bf;
typedef __attribute__((ext_vector_type(8))) float v8f;

union FragB { unsigned u[8]; v16bf v; };

__device__ __forceinline__ unsigned bf16pack(float a, float b) {
  unsigned ua = __float_as_uint(a), ub = __float_as_uint(b);
  ua = (ua + 0x7FFFu + ((ua >> 16) & 1u)) >> 16;
  ub = (ub + 0x7FFFu + ((ub >> 16) & 1u)) >> 16;
  return (ua & 0xFFFFu) | (ub << 16);
}

// monotone mapping float -> uint so atomicMax(u32) == float max
__device__ __forceinline__ unsigned fenc(float f) {
  unsigned u = __float_as_uint(f);
  return (u >> 31) ? ~u : (u | 0x80000000u);
}
__device__ __forceinline__ float fdec(unsigned e) {
  unsigned u = (e >> 31) ? (e & 0x7FFFFFFFu) : ~e;
  return __uint_as_float(u);
}

// reduce across the 16 lanes of a half-wave (xor masks stay inside the half)
__device__ __forceinline__ float halfsum(float v) {
  v += __shfl_xor(v, 1, 32);
  v += __shfl_xor(v, 2, 32);
  v += __shfl_xor(v, 4, 32);
  v += __shfl_xor(v, 8, 32);
  return v;
}

__device__ __forceinline__ void setFrag(FragB& f, uint4 a, uint4 b) {
  f.u[0] = a.x; f.u[1] = a.y; f.u[2] = a.z; f.u[3] = a.w;
  f.u[4] = b.x; f.u[5] = b.y; f.u[6] = b.z; f.u[7] = b.w;
}

__device__ __forceinline__ v8f wmma_bf16(const FragB& a, const FragB& b, v8f c) {
  return __builtin_amdgcn_wmma_f32_16x16x32_bf16(false, a.v, false, b.v, (short)0, c,
                                                 false, false);
}

// B fragment: prepacked so each lane reads 32 contiguous bytes (2 x b128)
__device__ __forceinline__ void loadB(const unsigned* __restrict__ pack, int tile,
                                      int lane, FragB& b) {
  const uint4* p = reinterpret_cast<const uint4*>(pack + (size_t)tile * 256 + lane * 8);
  setFrag(b, p[0], p[1]);
}

// -------------------- utility kernels --------------------

__global__ void zero_kernel(unsigned* p, long n) {
  long i = (long)blockIdx.x * blockDim.x + threadIdx.x;
  long stride = (long)gridDim.x * blockDim.x;
  for (; i < n; i += stride) p[i] = 0u;
}

// pack W[K][Nc] (f32 row-major) into WMMA-B bf16 fragments:
// out[tile*256 + lane*8 + j] = pack(W[k0][n], W[k0+1][n])
// tile = kt*NNT + nt ; n = nt*16 + lane%16 ; k0 = kt*32 + (j/4)*16 + (lane/16)*8 + 2(j%4)
__global__ void pack_kernel(const float* __restrict__ W, unsigned* __restrict__ out,
                            int Nc, int NNT, int total) {
  int i = blockIdx.x * blockDim.x + threadIdx.x;
  if (i >= total) return;
  int j = i & 7, lane = (i >> 3) & 31, tile = i >> 8;
  int nt = tile % NNT, kt = tile / NNT;
  int n = nt * 16 + (lane & 15);
  int k0 = kt * 32 + (j >> 2) * 16 + (lane >> 4) * 8 + 2 * (j & 3);
  out[i] = bf16pack(W[(size_t)k0 * Nc + n], W[(size_t)(k0 + 1) * Nc + n]);
}

// -------------------- stage 1: edge MLP + segment max --------------------

__global__ __launch_bounds__(128) void edge_kernel(
    const float* __restrict__ x, const float* __restrict__ edge_attr,
    const int* __restrict__ row, const int* __restrict__ col,
    const unsigned* __restrict__ packE, const float* __restrict__ b_e,
    const float* __restrict__ g_e, const float* __restrict__ be_e,
    const float* __restrict__ tp, float* __restrict__ e_out,
    unsigned* __restrict__ smax_enc) {
  __shared__ unsigned lds[4 * 16 * 192];  // 4 waves * 16 edges * 384 bf16 = 48KB
  const int lane = threadIdx.x & 31;
  const int wid = threadIdx.x >> 5;
  const int l15 = lane & 15, khalf = lane >> 4;
  const int ebase = (blockIdx.x * 4 + wid) * 16;
  unsigned* wl = lds + wid * 3072;

  // stage gathered [x_src | x_dst | edge_attr] rows as packed bf16
  for (int le = 0; le < 16; ++le) {
    int ei = ebase + le; if (ei >= N_EDGES) ei = N_EDGES - 1;
    int r = row[ei], c = col[ei];
    float4 v0 = reinterpret_cast<const float4*>(x + (size_t)r * DIM)[lane];
    float4 v1 = reinterpret_cast<const float4*>(x + (size_t)c * DIM)[lane];
    float4 v2 = reinterpret_cast<const float4*>(edge_attr + (size_t)ei * DIM)[lane];
    unsigned* p = wl + le * 192 + lane * 2;
    p[0]   = bf16pack(v0.x, v0.y); p[1]   = bf16pack(v0.z, v0.w);
    p[64]  = bf16pack(v1.x, v1.y); p[65]  = bf16pack(v1.z, v1.w);
    p[128] = bf16pack(v2.x, v2.y); p[129] = bf16pack(v2.z, v2.w);
  }
  __syncthreads();

  v8f acc[8];
#pragma unroll
  for (int nt = 0; nt < 8; ++nt) acc[nt] = (v8f){0.f,0.f,0.f,0.f,0.f,0.f,0.f,0.f};

  const unsigned* arow = wl + l15 * 192;
  for (int kt = 0; kt < 12; ++kt) {
    const uint4* ap = reinterpret_cast<const uint4*>(arow + kt * 16 + khalf * 4);
    FragB a; setFrag(a, ap[0], ap[2]);
#pragma unroll
    for (int nt = 0; nt < 8; ++nt) {
      FragB b; loadB(packE, kt * 8 + nt, lane, b);
      acc[nt] = wmma_bf16(a, b, acc[nt]);
    }
  }

  const float tval = *tp;
  float bev[8], gev[8], beev[8];
#pragma unroll
  for (int nt = 0; nt < 8; ++nt) {
    int n = nt * 16 + l15;
    bev[nt] = b_e[n]; gev[nt] = g_e[n]; beev[nt] = be_e[n];
  }
  // bias + relu
#pragma unroll
  for (int nt = 0; nt < 8; ++nt)
#pragma unroll
    for (int j = 0; j < 8; ++j) {
      float v = acc[nt][j] + bev[nt];
      acc[nt][j] = v > 0.f ? v : 0.f;
    }
  // per-edge-row LayerNorm stats (row m = khalf*8 + j, 128 chans across half-wave)
  float mu[8], rinv[8];
#pragma unroll
  for (int j = 0; j < 8; ++j) {
    float s1 = 0.f, s2 = 0.f;
#pragma unroll
    for (int nt = 0; nt < 8; ++nt) { float v = acc[nt][j]; s1 += v; s2 += v * v; }
    s1 = halfsum(s1); s2 = halfsum(s2);
    float m = s1 * (1.f / DIM);
    mu[j] = m;
    rinv[j] = rsqrtf(s2 * (1.f / DIM) - m * m + 1e-5f);
  }
#pragma unroll
  for (int j = 0; j < 8; ++j) {
    int ei = ebase + khalf * 8 + j;
    bool valid = ei < N_EDGES;
    int eic = valid ? ei : N_EDGES - 1;
    int r = row[eic], cd = col[eic];
#pragma unroll
    for (int nt = 0; nt < 8; ++nt) {
      int n = nt * 16 + l15;
      float eav = edge_attr[(size_t)eic * DIM + n];
      float ev = (acc[nt][j] - mu[j]) * rinv[j] * gev[nt] + beev[nt] + eav;
      if (valid) e_out[(size_t)ei * DIM + n] = ev;
      float msg = x[(size_t)r * DIM + n] + ev;
      msg = (msg > 0.f ? msg : 0.f) + 1e-7f;
      if (valid) atomicMax(smax_enc + (size_t)cd * DIM + n, fenc(msg * tval));
    }
  }
}

// -------------------- stage 2: softmax accumulation --------------------

__global__ void softmax_accum_kernel(
    const float* __restrict__ x, const float* __restrict__ e_out,
    const int* __restrict__ row, const int* __restrict__ col,
    const unsigned* __restrict__ smax_enc, const float* __restrict__ tp,
    float* __restrict__ num, float* __restrict__ den) {
  long i = (long)blockIdx.x * blockDim.x + threadIdx.x;
  if (i >= (long)N_EDGES * DIM) return;
  int e = (int)(i >> 7), n = (int)(i & 127);
  int r = row[e], c = col[e];
  float msg = x[(size_t)r * DIM + n] + e_out[i];
  msg = fmaxf(msg, 0.f) + 1e-7f;
  float s = msg * (*tp);
  float ex = __expf(s - fdec(smax_enc[(size_t)c * DIM + n]));
  atomicAdd(den + (size_t)c * DIM + n, ex);
  atomicAdd(num + (size_t)c * DIM + n, msg * ex);
}

// -------------------- stage 3: node MLP layer 1 + BN stats --------------------

__global__ __launch_bounds__(128) void node_mlp1_kernel(
    const float* __restrict__ x, const float* __restrict__ num,
    const float* __restrict__ den, const unsigned* __restrict__ packM1,
    const float* __restrict__ b_m1, float* __restrict__ h1,
    float* __restrict__ bnsum, float* __restrict__ bnsumsq) {
  __shared__ unsigned lds[4 * 16 * 64];  // 16KB
  const int lane = threadIdx.x & 31;
  const int wid = threadIdx.x >> 5;
  const int l15 = lane & 15, khalf = lane >> 4;
  const int nbase = (blockIdx.x * 4 + wid) * 16;
  const int ybase = blockIdx.y * 8;  // output ntile offset (256 chans = 16 ntiles)
  unsigned* wl = lds + wid * 1024;

  for (int le = 0; le < 16; ++le) {
    int nd = nbase + le; if (nd >= N_NODES) nd = N_NODES - 1;
    float4 xv = reinterpret_cast<const float4*>(x + (size_t)nd * DIM)[lane];
    float4 nv = reinterpret_cast<const float4*>(num + (size_t)nd * DIM)[lane];
    float4 dv = reinterpret_cast<const float4*>(den + (size_t)nd * DIM)[lane];
    float a0 = xv.x + (dv.x > 0.f ? nv.x / dv.x : 0.f);
    float a1 = xv.y + (dv.y > 0.f ? nv.y / dv.y : 0.f);
    float a2 = xv.z + (dv.z > 0.f ? nv.z / dv.z : 0.f);
    float a3 = xv.w + (dv.w > 0.f ? nv.w / dv.w : 0.f);
    unsigned* p = wl + le * 64 + lane * 2;
    p[0] = bf16pack(a0, a1); p[1] = bf16pack(a2, a3);
  }
  __syncthreads();

  v8f acc[8];
#pragma unroll
  for (int nt = 0; nt < 8; ++nt) acc[nt] = (v8f){0.f,0.f,0.f,0.f,0.f,0.f,0.f,0.f};
  const unsigned* arow = wl + l15 * 64;
  for (int kt = 0; kt < 4; ++kt) {
    const uint4* ap = reinterpret_cast<const uint4*>(arow + kt * 16 + khalf * 4);
    FragB a; setFrag(a, ap[0], ap[2]);
#pragma unroll
    for (int nt = 0; nt < 8; ++nt) {
      FragB b; loadB(packM1, kt * 16 + ybase + nt, lane, b);
      acc[nt] = wmma_bf16(a, b, acc[nt]);
    }
  }

#pragma unroll
  for (int nt = 0; nt < 8; ++nt) {
    int n = (ybase + nt) * 16 + l15;
    float bm = b_m1[n];
    float ps = 0.f, pss = 0.f;
#pragma unroll
    for (int j = 0; j < 8; ++j) {
      int nd = nbase + khalf * 8 + j;
      float v = acc[nt][j] + bm;
      bool valid = nd < N_NODES;
      if (valid) h1[(size_t)nd * 256 + n] = v;
      float vv = valid ? v : 0.f;
      ps += vv; pss += vv * vv;
    }
    ps += __shfl_xor(ps, 16, 32);
    pss += __shfl_xor(pss, 16, 32);
    if (khalf == 0) { atomicAdd(bnsum + n, ps); atomicAdd(bnsumsq + n, pss); }
  }
}

__global__ void bn_finalize_kernel(const float* __restrict__ bnsum,
                                   const float* __restrict__ bnsumsq,
                                   const float* __restrict__ g_bn,
                                   const float* __restrict__ b_bn,
                                   float* __restrict__ bnscale,
                                   float* __restrict__ bnshift) {
  int ch = threadIdx.x;
  if (ch < 256) {
    float mu = bnsum[ch] * (1.f / N_NODES);
    float var = bnsumsq[ch] * (1.f / N_NODES) - mu * mu;
    float sc = g_bn[ch] * rsqrtf(var + 1e-5f);
    bnscale[ch] = sc;
    bnshift[ch] = b_bn[ch] - mu * sc;
  }
}

// -------------------- stage 4: node MLP layer 2 + MessageNorm + LN ----------

__global__ __launch_bounds__(128) void node_mlp2_kernel(
    const float* __restrict__ x, const float* __restrict__ h1,
    const float* __restrict__ bnscale, const float* __restrict__ bnshift,
    const unsigned* __restrict__ packM2, const float* __restrict__ b_m2,
    const float* __restrict__ scalep, const float* __restrict__ g_n,
    const float* __restrict__ b_n, float* __restrict__ x_out) {
  __shared__ unsigned lds[4 * 16 * 128];  // 32KB
  const int lane = threadIdx.x & 31;
  const int wid = threadIdx.x >> 5;
  const int l15 = lane & 15, khalf = lane >> 4;
  const int nbase = (blockIdx.x * 4 + wid) * 16;
  unsigned* wl = lds + wid * 2048;

  float4 sc0 = reinterpret_cast<const float4*>(bnscale)[lane * 2];
  float4 sc1 = reinterpret_cast<const float4*>(bnscale)[lane * 2 + 1];
  float4 sh0 = reinterpret_cast<const float4*>(bnshift)[lane * 2];
  float4 sh1 = reinterpret_cast<const float4*>(bnshift)[lane * 2 + 1];

  for (int le = 0; le < 16; ++le) {
    int nd = nbase + le; if (nd >= N_NODES) nd = N_NODES - 1;
    const float4* hp = reinterpret_cast<const float4*>(h1 + (size_t)nd * 256);
    float4 a = hp[lane * 2], b = hp[lane * 2 + 1];
    float v0 = fmaxf(a.x * sc0.x + sh0.x, 0.f), v1 = fmaxf(a.y * sc0.y + sh0.y, 0.f);
    float v2 = fmaxf(a.z * sc0.z + sh0.z, 0.f), v3 = fmaxf(a.w * sc0.w + sh0.w, 0.f);
    float v4 = fmaxf(b.x * sc1.x + sh1.x, 0.f), v5 = fmaxf(b.y * sc1.y + sh1.y, 0.f);
    float v6 = fmaxf(b.z * sc1.z + sh1.z, 0.f), v7 = fmaxf(b.w * sc1.w + sh1.w, 0.f);
    unsigned* p = wl + le * 128 + lane * 4;
    p[0] = bf16pack(v0, v1); p[1] = bf16pack(v2, v3);
    p[2] = bf16pack(v4, v5); p[3] = bf16pack(v6, v7);
  }
  __syncthreads();

  v8f acc[8];
#pragma unroll
  for (int nt = 0; nt < 8; ++nt) acc[nt] = (v8f){0.f,0.f,0.f,0.f,0.f,0.f,0.f,0.f};
  const unsigned* arow = wl + l15 * 128;
  for (int kt = 0; kt < 8; ++kt) {
    const uint4* ap = reinterpret_cast<const uint4*>(arow + kt * 16 + khalf * 4);
    FragB a; setFrag(a, ap[0], ap[2]);
#pragma unroll
    for (int nt = 0; nt < 8; ++nt) {
      FragB b; loadB(packM2, kt * 8 + nt, lane, b);
      acc[nt] = wmma_bf16(a, b, acc[nt]);
    }
  }

  const float smn = *scalep;
  float bm2[8], gn[8], bn[8];
#pragma unroll
  for (int nt = 0; nt < 8; ++nt) {
    int n = nt * 16 + l15;
    bm2[nt] = b_m2[n]; gn[nt] = g_n[n]; bn[nt] = b_n[n];
  }
#pragma unroll
  for (int nt = 0; nt < 8; ++nt)
#pragma unroll
    for (int j = 0; j < 8; ++j) acc[nt][j] += bm2[nt];

  int ndj[8]; bool val[8];
#pragma unroll
  for (int j = 0; j < 8; ++j) {
    int nd = nbase + khalf * 8 + j;
    val[j] = nd < N_NODES;
    ndj[j] = val[j] ? nd : N_NODES - 1;
  }
  float xvv[8][8];
#pragma unroll
  for (int nt = 0; nt < 8; ++nt) {
    int n = nt * 16 + l15;
#pragma unroll
    for (int j = 0; j < 8; ++j) xvv[nt][j] = x[(size_t)ndj[j] * DIM + n];
  }
  // MessageNorm factor per row
  float fac[8];
#pragma unroll
  for (int j = 0; j < 8; ++j) {
    float hs = 0.f, xs = 0.f;
#pragma unroll
    for (int nt = 0; nt < 8; ++nt) {
      float h = acc[nt][j]; hs += h * h;
      float xw = xvv[nt][j]; xs += xw * xw;
    }
    hs = halfsum(hs); xs = halfsum(xs);
    fac[j] = sqrtf(xs) * smn / fmaxf(sqrtf(hs), 1e-12f);
  }
  // residual t = x + h * fac
#pragma unroll
  for (int nt = 0; nt < 8; ++nt)
#pragma unroll
    for (int j = 0; j < 8; ++j) acc[nt][j] = xvv[nt][j] + acc[nt][j] * fac[j];
  // node LayerNorm
  float mu[8], rinv[8];
#pragma unroll
  for (int j = 0; j < 8; ++j) {
    float s1 = 0.f, s2 = 0.f;
#pragma unroll
    for (int nt = 0; nt < 8; ++nt) { float v = acc[nt][j]; s1 += v; s2 += v * v; }
    s1 = halfsum(s1); s2 = halfsum(s2);
    float m = s1 * (1.f / DIM);
    mu[j] = m;
    rinv[j] = rsqrtf(s2 * (1.f / DIM) - m * m + 1e-5f);
  }
#pragma unroll
  for (int j = 0; j < 8; ++j)
#pragma unroll
    for (int nt = 0; nt < 8; ++nt) {
      int n = nt * 16 + l15;
      float o = (acc[nt][j] - mu[j]) * rinv[j] * gn[nt] + bn[nt];
      if (val[j]) x_out[(size_t)ndj[j] * DIM + n] = o;
    }
}

// -------------------- launcher --------------------

extern "C" void kernel_launch(void* const* d_in, const int* in_sizes, int n_in,
                              void* d_out, int out_size, void* d_ws, size_t ws_size,
                              hipStream_t stream) {
  const float* x         = (const float*)d_in[0];
  const float* edge_attr = (const float*)d_in[1];
  const int*   ei        = (const int*)d_in[2];
  const int*   row       = ei;
  const int*   col       = ei + N_EDGES;
  const float* W_e  = (const float*)d_in[3];
  const float* b_e  = (const float*)d_in[4];
  const float* g_e  = (const float*)d_in[5];
  const float* be_e = (const float*)d_in[6];
  const float* tp   = (const float*)d_in[7];
  const float* W_m1 = (const float*)d_in[8];
  const float* b_m1 = (const float*)d_in[9];
  const float* g_bn = (const float*)d_in[10];
  const float* b_bn = (const float*)d_in[11];
  const float* W_m2 = (const float*)d_in[12];
  const float* b_m2 = (const float*)d_in[13];
  const float* scalep = (const float*)d_in[14];
  const float* g_n  = (const float*)d_in[15];
  const float* b_n  = (const float*)d_in[16];

  float* x_out = (float*)d_out;
  float* e_out = (float*)d_out + (size_t)N_NODES * DIM;

  const size_t NM = (size_t)N_NODES * DIM;
  unsigned* ws      = (unsigned*)d_ws;
  unsigned* smax    = ws;
  float*    num     = (float*)(ws + NM);
  float*    den     = (float*)(ws + 2 * NM);
  float*    h1      = (float*)(ws + 3 * NM);
  float*    bnsum   = (float*)(ws + 3 * NM + (size_t)N_NODES * 256);
  float*    bnsumsq = bnsum + 256;
  float*    bnscale = bnsum + 512;
  float*    bnshift = bnsum + 768;
  unsigned* packE   = (unsigned*)(bnsum + 1024);
  unsigned* packM1  = packE + 24576;
  unsigned* packM2  = packM1 + 16384;

  zero_kernel<<<2048, 256, 0, stream>>>(smax, (long)(3 * NM));
  zero_kernel<<<2, 256, 0, stream>>>((unsigned*)bnsum, 1024);

  pack_kernel<<<(24576 + 255) / 256, 256, 0, stream>>>(W_e, packE, 128, 8, 24576);
  pack_kernel<<<(16384 + 255) / 256, 256, 0, stream>>>(W_m1, packM1, 256, 16, 16384);
  pack_kernel<<<(16384 + 255) / 256, 256, 0, stream>>>(W_m2, packM2, 128, 8, 16384);

  edge_kernel<<<(N_EDGES + 63) / 64, 128, 0, stream>>>(
      x, edge_attr, row, col, packE, b_e, g_e, be_e, tp, e_out, smax);

  long etot = (long)N_EDGES * DIM;
  softmax_accum_kernel<<<(unsigned)((etot + 255) / 256), 256, 0, stream>>>(
      x, e_out, row, col, smax, tp, num, den);

  dim3 g1((N_NODES + 63) / 64, 2);
  node_mlp1_kernel<<<g1, 128, 0, stream>>>(x, num, den, packM1, b_m1, h1, bnsum,
                                           bnsumsq);

  bn_finalize_kernel<<<1, 256, 0, stream>>>(bnsum, bnsumsq, g_bn, b_bn, bnscale,
                                            bnshift);

  node_mlp2_kernel<<<(N_NODES + 63) / 64, 128, 0, stream>>>(
      x, h1, bnscale, bnshift, packM2, b_m2, scalep, g_n, b_n, x_out);
}
